// InputFeedRNNDecoder_14886356648302
// MI455X (gfx1250) — matrix operand order repaired and checked by reading, not compiled, not run on hardware
//
#include <hip/hip_runtime.h>
#include <hip/hip_bf16.h>

// ---------------------------------------------------------------------------
// InputFeedRNNDecoder for MI455X (gfx1250): bf16 WMMA GEMMs, fp32 accum.
// T=64 B=32 S=64 H=1024 E=1024 V=32000 L=2
// ---------------------------------------------------------------------------

typedef __attribute__((ext_vector_type(16))) __bf16 v16bf;
typedef __attribute__((ext_vector_type(8)))  float  v8f;
typedef __hip_bfloat16 bf16_t;

union Frag16 { v16bf v; uint4 q[2]; };

// A-matrix fragment (16x32 bf16, M x K). Lane m(0..15)/m+16 holds row m,
// K chunks {k0+half*8 .. +7} and {k0+16+half*8 .. +7} (per ISA layout table).
__device__ __forceinline__ v16bf load_afrag(const bf16_t* X, int ldx, int mbase,
                                            int k0, int lane) {
  int m    = mbase + (lane & 15);
  int half = lane >> 4;
  const bf16_t* p = X + (size_t)m * ldx + k0 + half * 8;
  Frag16 f;
  f.q[0] = *(const uint4*)p;        // K = k0+half*8   .. +7
  f.q[1] = *(const uint4*)(p + 16); // K = k0+16+half*8 .. +7
  return f.v;
}

// B-matrix fragment (32x16 bf16, K x N) built from W stored [N,K] row-major
// (B[k][n] = W[n][k]). Lane n holds K chunks {k0+half*8 .. +7} and {k0+16+half*8 .. +7}.
__device__ __forceinline__ v16bf load_bfrag(const bf16_t* W, int ldw, int nbase,
                                            int k0, int lane) {
  int n    = nbase + (lane & 15);
  int half = lane >> 4;
  const bf16_t* p = W + (size_t)n * ldw + k0 + half * 8;
  Frag16 f;
  f.q[0] = *(const uint4*)p;
  f.q[1] = *(const uint4*)(p + 16);
  return f.v;
}

__device__ __forceinline__ v8f wmma_bf16(v16bf a, v16bf b, v8f c) {
  return __builtin_amdgcn_wmma_f32_16x16x32_bf16(false, a, false, b,
                                                 (short)0, c, false, false);
}

__device__ __forceinline__ float sigf(float x) { return 1.f / (1.f + expf(-x)); }

// ---------------------------------------------------------------------------
// Prep kernels
// ---------------------------------------------------------------------------

// f32 [rows,cols] -> bf16 packed at dst[r*dstStride + dstOff + c]
__global__ void cvt_pack_kernel(bf16_t* dst, const float* src, int rows, int cols,
                                int dstStride, int dstOff) {
  int n = rows * cols;
  for (int i = blockIdx.x * blockDim.x + threadIdx.x; i < n;
       i += gridDim.x * blockDim.x) {
    int r = i / cols, c = i % cols;
    dst[(size_t)r * dstStride + dstOff + c] = __float2bfloat16(src[i]);
  }
}

__global__ void add_bias_kernel(float* dst, const float* a, const float* b, int n) {
  int i = blockIdx.x * blockDim.x + threadIdx.x;
  if (i < n) dst[i] = a[i] + b[i];
}

// Copy initial cell states, convert initial hidden states to bf16, zero feed.
__global__ void init_state_kernel(float* c0s, float* c1s, bf16_t* h0bf, bf16_t* h1bf,
                                  bf16_t* feedbf, const float* hin, const float* cin) {
  int i = blockIdx.x * blockDim.x + threadIdx.x;
  if (i < 32 * 1024) {
    c0s[i]   = cin[i];
    c1s[i]   = cin[32 * 1024 + i];
    h0bf[i]  = __float2bfloat16(hin[i]);
    h1bf[i]  = __float2bfloat16(hin[32 * 1024 + i]);
    feedbf[i] = __float2bfloat16(0.f);
  }
}

// ---------------------------------------------------------------------------
// Per-step: assemble x0 = [emb_t | feed | h0_prev] (bf16 [32,3072]) and the
// h1_prev half of x1 (bf16 [32,2048], cols 1024..2047).
// ---------------------------------------------------------------------------
__global__ void assemble_kernel(bf16_t* x0, bf16_t* x1, const float* emb,
                                const int* ids, const bf16_t* feedbf,
                                const bf16_t* h0bf, const bf16_t* h1bf, int t) {
  int idx = blockIdx.x * 256 + threadIdx.x; // 512 blocks * 256 = 131072
  if (idx < 32 * 3072) {
    int m = idx / 3072, k = idx % 3072;
    if (k < 1024) {
      int id = ids[t * 32 + m];
      x0[idx] = __float2bfloat16(emb[(size_t)id * 1024 + k]);
    } else if (k < 2048) {
      x0[idx] = feedbf[m * 1024 + (k - 1024)];
    } else {
      x0[idx] = h0bf[m * 1024 + (k - 2048)];
    }
  } else {
    int j = idx - 32 * 3072; // < 32*1024
    int m = j >> 10, k = j & 1023;
    x1[m * 2048 + 1024 + k] = h1bf[j];
  }
}

// ---------------------------------------------------------------------------
// Fused LSTM cell: gates = X[32,K] @ Wcat[4096,K]^T + bias, then elementwise.
// Block b owns hidden cols [b*64, b*64+64); 8 waves: wave w -> gate g=w/2,
// n-offset (w&1)*32. Each wave does a 32x32 tile = 4 WMMA accumulators.
// Gates meet in LDS for the pointwise i/f/g/o combine.
// ---------------------------------------------------------------------------
__global__ void lstm_kernel(const bf16_t* X, int K, const bf16_t* W,
                            const float* bias, float* c, bf16_t* hbf,
                            bf16_t* xnext, int xnextStride, int xnextOff) {
  __shared__ float gbuf[4][32][64]; // 32 KB
  const int lane = threadIdx.x & 31;
  const int wave = threadIdx.x >> 5;
  const int g    = wave >> 1;
  const int noff = (wave & 1) * 32;
  const int hb   = blockIdx.x * 64;
  const int ncol = g * 1024 + hb + noff; // row base in W (gate output index)

  v8f a00 = {}, a01 = {}, a10 = {}, a11 = {};
  for (int k0 = 0; k0 < K; k0 += 32) {
    v16bf fa0 = load_afrag(X, K, 0, k0, lane);
    v16bf fa1 = load_afrag(X, K, 16, k0, lane);
    v16bf fb0 = load_bfrag(W, K, ncol, k0, lane);
    v16bf fb1 = load_bfrag(W, K, ncol + 16, k0, lane);
    a00 = wmma_bf16(fa0, fb0, a00);
    a01 = wmma_bf16(fa0, fb1, a01);
    a10 = wmma_bf16(fa1, fb0, a10);
    a11 = wmma_bf16(fa1, fb1, a11);
  }

  // C/D layout: VGPR r -> M = r + 8*(lane>=16); N = lane&15.
  const int half = lane >> 4, nl = lane & 15;
#pragma unroll
  for (int r = 0; r < 8; ++r) {
    int m0 = half * 8 + r, m1 = 16 + half * 8 + r;
    gbuf[g][m0][noff + nl]      = a00[r];
    gbuf[g][m0][noff + 16 + nl] = a01[r];
    gbuf[g][m1][noff + nl]      = a10[r];
    gbuf[g][m1][noff + 16 + nl] = a11[r];
  }
  __syncthreads();

  for (int e = threadIdx.x; e < 32 * 64; e += 256) {
    int m = e >> 6, j = e & 63;
    int n = hb + j;
    float iv = gbuf[0][m][j] + bias[n];
    float fv = gbuf[1][m][j] + bias[1024 + n];
    float gv = gbuf[2][m][j] + bias[2048 + n];
    float ov = gbuf[3][m][j] + bias[3072 + n];
    float cp = c[m * 1024 + n];
    float cn = sigf(fv) * cp + sigf(iv) * tanhf(gv);
    float hn = sigf(ov) * tanhf(cn);
    c[m * 1024 + n]   = cn;
    hbf[m * 1024 + n] = __float2bfloat16(hn);
    if (xnext) xnext[m * xnextStride + xnextOff + n] = __float2bfloat16(hn);
  }
}

// ---------------------------------------------------------------------------
// Plain GEMM Y = X[32,K] @ W[1024,K]^T, optional tanh, fp32 and/or bf16 out.
// 8 blocks, 8 waves each; wave owns a 32x16 output strip (2 accumulators).
// ---------------------------------------------------------------------------
__global__ void gemm_kernel(const bf16_t* X, int K, const bf16_t* W,
                            float* yf, int ystride, bf16_t* ybf, int do_tanh) {
  const int lane = threadIdx.x & 31;
  const int wave = threadIdx.x >> 5;
  const int nb   = blockIdx.x * 128 + wave * 16;

  v8f acc0 = {}, acc1 = {};
  for (int k0 = 0; k0 < K; k0 += 32) {
    v16bf fa0 = load_afrag(X, K, 0, k0, lane);
    v16bf fa1 = load_afrag(X, K, 16, k0, lane);
    v16bf fb  = load_bfrag(W, K, nb, k0, lane);
    acc0 = wmma_bf16(fa0, fb, acc0);
    acc1 = wmma_bf16(fa1, fb, acc1);
  }

  const int half = lane >> 4, nl = lane & 15, n = nb + nl;
#pragma unroll
  for (int r = 0; r < 8; ++r) {
    float v0 = acc0[r], v1 = acc1[r];
    if (do_tanh) { v0 = tanhf(v0); v1 = tanhf(v1); }
    int m0 = half * 8 + r, m1 = 16 + half * 8 + r;
    if (yf)  { yf[m0 * ystride + n] = v0; yf[m1 * ystride + n] = v1; }
    if (ybf) { ybf[m0 * 1024 + n] = __float2bfloat16(v0);
               ybf[m1 * 1024 + n] = __float2bfloat16(v1); }
  }
}

// ---------------------------------------------------------------------------
// Attention (general): scores = q . ctx, masked softmax, c_vec = attn . ctx.
// One block per batch row. Writes attn row to d_out and x2 = [c_vec | h1] bf16.
// ---------------------------------------------------------------------------
__global__ void attn_kernel(const float* q, const float* context, const int* lens,
                            const bf16_t* h1bf, bf16_t* x2, float* dout, int t) {
  __shared__ float qs[1024];
  __shared__ float part[64][4];
  __shared__ float attn_s[64];
  __shared__ float inv_sum;
  const int b = blockIdx.x, tid = threadIdx.x;

  for (int i = tid; i < 1024; i += 256) qs[i] = q[b * 1024 + i];
  __syncthreads();

  { // scores: 4 threads per source position
    int s = tid >> 2, quad = tid & 3;
    const float* crow = context + ((size_t)(s * 32 + b)) * 1024;
    float p = 0.f;
    for (int h = quad * 256; h < quad * 256 + 256; ++h) p += qs[h] * crow[h];
    part[s][quad] = p;
  }
  __syncthreads();

  if (tid < 64) {
    int len = lens[b];
    len = len < 1 ? 1 : (len > 64 ? 64 : len);
    float sc = part[tid][0] + part[tid][1] + part[tid][2] + part[tid][3];
    if (tid >= len) sc = -3.402823466e38f;
    attn_s[tid] = sc;
  }
  __syncthreads();
  if (tid == 0) { // softmax over 64 (cheap serial)
    float mx = attn_s[0];
    for (int s = 1; s < 64; ++s) mx = fmaxf(mx, attn_s[s]);
    float sum = 0.f;
    for (int s = 0; s < 64; ++s) { float e = expf(attn_s[s] - mx); attn_s[s] = e; sum += e; }
    inv_sum = 1.f / sum;
  }
  __syncthreads();
  if (tid < 64) {
    float a = attn_s[tid] * inv_sum;
    attn_s[tid] = a;
    dout[64 * 32 * 1024 + ((size_t)(t * 32 + b)) * 64 + tid] = a; // attns[t,b,s]
  }
  __syncthreads();

  for (int h = tid; h < 1024; h += 256) {
    float acc = 0.f;
    for (int s = 0; s < 64; ++s)
      acc += attn_s[s] * context[((size_t)(s * 32 + b)) * 1024 + h];
    x2[b * 2048 + h]        = __float2bfloat16(acc);         // c_vec
    x2[b * 2048 + 1024 + h] = h1bf[b * 1024 + h];            // rnn_out
  }
}

// ---------------------------------------------------------------------------
// Host side
// ---------------------------------------------------------------------------
// Workspace layout (bytes)
static const size_t WCAT0_OFF = 0;                         // [4096,3072] bf16
static const size_t WCAT1_OFF = WCAT0_OFF + 4096ull*3072*2; // [4096,2048] bf16
static const size_t WIN_OFF   = WCAT1_OFF + 4096ull*2048*2; // [1024,1024] bf16
static const size_t WOUT_OFF  = WIN_OFF   + 1024ull*1024*2; // [1024,2048] bf16
static const size_t BIAS0_OFF = WOUT_OFF  + 1024ull*2048*2; // [4096] f32
static const size_t BIAS1_OFF = BIAS0_OFF + 4096ull*4;
static const size_t C0_OFF    = BIAS1_OFF + 4096ull*4;      // [32,1024] f32
static const size_t C1_OFF    = C0_OFF    + 32ull*1024*4;
static const size_t H0BF_OFF  = C1_OFF    + 32ull*1024*4;   // [32,1024] bf16
static const size_t H1BF_OFF  = H0BF_OFF  + 32ull*1024*2;
static const size_t FEED_OFF  = H1BF_OFF  + 32ull*1024*2;
static const size_t X0_OFF    = FEED_OFF  + 32ull*1024*2;   // [32,3072] bf16
static const size_t X1_OFF    = X0_OFF    + 32ull*3072*2;   // [32,2048] bf16
static const size_t X2_OFF    = X1_OFF    + 32ull*2048*2;   // [32,2048] bf16
static const size_t Q_OFF     = X2_OFF    + 32ull*2048*2;   // [32,1024] f32

extern "C" void kernel_launch(void* const* d_in, const int* in_sizes, int n_in,
                              void* d_out, int out_size, void* d_ws, size_t ws_size,
                              hipStream_t stream) {
  (void)in_sizes; (void)n_in; (void)out_size; (void)ws_size;

  const int*   input_ids = (const int*)  d_in[0];
  const float* context   = (const float*)d_in[1];
  const int*   ctx_len   = (const int*)  d_in[2];
  const float* h0in      = (const float*)d_in[3];
  const float* c0in      = (const float*)d_in[4];
  const float* embedding = (const float*)d_in[5];
  const float* W_ih0     = (const float*)d_in[6];
  const float* W_hh0     = (const float*)d_in[7];
  const float* b_ih0     = (const float*)d_in[8];
  const float* b_hh0     = (const float*)d_in[9];
  const float* W_ih1     = (const float*)d_in[10];
  const float* W_hh1     = (const float*)d_in[11];
  const float* b_ih1     = (const float*)d_in[12];
  const float* b_hh1     = (const float*)d_in[13];
  const float* W_in      = (const float*)d_in[14];
  const float* W_out     = (const float*)d_in[15];

  char* ws = (char*)d_ws;
  bf16_t* Wcat0  = (bf16_t*)(ws + WCAT0_OFF);
  bf16_t* Wcat1  = (bf16_t*)(ws + WCAT1_OFF);
  bf16_t* Winbf  = (bf16_t*)(ws + WIN_OFF);
  bf16_t* Woutbf = (bf16_t*)(ws + WOUT_OFF);
  float*  bias0  = (float*) (ws + BIAS0_OFF);
  float*  bias1  = (float*) (ws + BIAS1_OFF);
  float*  c0s    = (float*) (ws + C0_OFF);
  float*  c1s    = (float*) (ws + C1_OFF);
  bf16_t* h0bf   = (bf16_t*)(ws + H0BF_OFF);
  bf16_t* h1bf   = (bf16_t*)(ws + H1BF_OFF);
  bf16_t* feedbf = (bf16_t*)(ws + FEED_OFF);
  bf16_t* x0     = (bf16_t*)(ws + X0_OFF);
  bf16_t* x1     = (bf16_t*)(ws + X1_OFF);
  bf16_t* x2     = (bf16_t*)(ws + X2_OFF);
  float*  qf32   = (float*) (ws + Q_OFF);
  float*  dout   = (float*)d_out;

  // ---- one-time prep: bf16 weight packs, combined biases, state init ----
  cvt_pack_kernel<<<2048, 256, 0, stream>>>(Wcat0, W_ih0, 4096, 2048, 3072, 0);
  cvt_pack_kernel<<<2048, 256, 0, stream>>>(Wcat0, W_hh0, 4096, 1024, 3072, 2048);
  cvt_pack_kernel<<<2048, 256, 0, stream>>>(Wcat1, W_ih1, 4096, 1024, 2048, 0);
  cvt_pack_kernel<<<2048, 256, 0, stream>>>(Wcat1, W_hh1, 4096, 1024, 2048, 1024);
  cvt_pack_kernel<<<1024, 256, 0, stream>>>(Winbf,  W_in,  1024, 1024, 1024, 0);
  cvt_pack_kernel<<<1024, 256, 0, stream>>>(Woutbf, W_out, 1024, 2048, 2048, 0);
  add_bias_kernel<<<16, 256, 0, stream>>>(bias0, b_ih0, b_hh0, 4096);
  add_bias_kernel<<<16, 256, 0, stream>>>(bias1, b_ih1, b_hh1, 4096);
  init_state_kernel<<<128, 256, 0, stream>>>(c0s, c1s, h0bf, h1bf, feedbf, h0in, c0in);

  // ---- sequential decode, stream-ordered (graph-capture friendly) ----
  for (int t = 0; t < 64; ++t) {
    assemble_kernel<<<512, 256, 0, stream>>>(x0, x1, embedding, input_ids,
                                             feedbf, h0bf, h1bf, t);
    // Layer 0: x0[32,3072] -> gates[32,4096]; writes h0bf and x1 cols 0..1023
    lstm_kernel<<<16, 256, 0, stream>>>(x0, 3072, Wcat0, bias0, c0s, h0bf,
                                        x1, 2048, 0);
    // Layer 1: x1[32,2048] -> gates[32,4096]; writes h1bf
    lstm_kernel<<<16, 256, 0, stream>>>(x1, 2048, Wcat1, bias1, c1s, h1bf,
                                        (bf16_t*)nullptr, 0, 0);
    // q = h1 @ W_in^T
    gemm_kernel<<<8, 256, 0, stream>>>(h1bf, 1024, Winbf, qf32, 1024,
                                       (bf16_t*)nullptr, 0);
    // masked softmax attention; builds x2 = [c_vec | h1], writes attns[t]
    attn_kernel<<<32, 256, 0, stream>>>(qf32, context, ctx_len, h1bf, x2, dout, t);
    // attn_h = tanh(x2 @ W_out^T) -> outputs[t] (f32) and feedbf (bf16)
    gemm_kernel<<<8, 256, 0, stream>>>(x2, 2048, Woutbf,
                                       dout + (size_t)t * 32 * 1024, 1024,
                                       feedbf, 1);
  }
}
